// MinConv2dLSTMCell_80075370267236
// MI455X (gfx1250) — compile-verified, matrix operand order checked by reading, not run on hardware
//
#include <hip/hip_runtime.h>

typedef __attribute__((ext_vector_type(16))) _Float16 v16h;
typedef __attribute__((ext_vector_type(8)))  float    v8f;

#define KB      18      // 576 / 32
#define S_TOTAL 32
#define SCHUNK  8
#define HID     64
#define HDIM    64
#define WDIM    64
#define CIN     64
#define KTOT    576     // CIN * 9

__global__ __launch_bounds__(256)
void minconv_gru_kernel(const float* __restrict__ x,     // (4,32,64,64,64)
                        const float* __restrict__ h0,    // (4,1,64,64,64)
                        const float* __restrict__ Wc,    // (128,64,3,3) == (128,576)
                        const float* __restrict__ bc,    // (128)
                        float* __restrict__ out,         // (4,32,64,64,64)
                        float* __restrict__ outLast)     // (4,1,64,64,64)
{
    extern __shared__ char smem[];
    _Float16* Blds   = (_Float16*)smem;            // 18*32*16 halves = 18 KB (B fragments)
    float*    convOut = (float*)(smem + 18432);    // 8*128*16 f32    = 64 KB (chunk of conv out)

    const int tid  = threadIdx.x;
    const int wave = tid >> 5;
    const int lane = tid & 31;
    const int hi   = lane >> 4;     // half-wave select
    const int lmod = lane & 15;

    const int b    = blockIdx.x >> 8;
    const int tile = blockIdx.x & 255;
    const int row  = tile >> 2;          // h position
    const int w0   = (tile & 3) << 4;    // 16-wide column tile base

    // ---- A fragments: this wave's 16x576 weight slice, resident in VGPRs ----
    const int mBase = wave << 4;
    v16h afrag[KB];
    #pragma unroll
    for (int kb = 0; kb < KB; ++kb) {
        v16h t;
        #pragma unroll
        for (int i = 0; i < 16; ++i) {
            int k = kb * 32 + ((i >> 3) << 4) + (hi << 3) + (i & 7);   // ISA 16-bit A layout
            t[i] = (_Float16)Wc[(mBase + lmod) * KTOT + k];
        }
        afrag[kb] = t;
    }
    // bias per accumulator row (C/D layout: m = r + 8*hi, n = lmod)
    float bias[8];
    #pragma unroll
    for (int r = 0; r < 8; ++r) bias[r] = bc[mBase + r + (hi << 3)];

    // ---- scan carry: each thread owns 4 (channel, pixel) pairs ----
    float hcar[4];
    #pragma unroll
    for (int j = 0; j < 4; ++j) {
        int p = tid + j * 256;
        int c = p >> 4, n = p & 15;
        hcar[j] = h0[((b * HID + c) * HDIM + row) * WDIM + (w0 + n)];
    }

    const float* xb = x + (size_t)b * S_TOTAL * CIN * HDIM * WDIM;

    for (int chunk = 0; chunk < 4; ++chunk) {
        for (int sL = 0; sL < SCHUNK; ++sL) {
            const int s = chunk * SCHUNK + sL;
            __syncthreads();   // previous B fragments fully consumed
            // ---- stage im2col patch, pre-swizzled to B-fragment order (36 halves/thread) ----
            const float* xs = xb + (size_t)s * CIN * HDIM * WDIM;
            for (int q = tid * 36, e = q + 36; q < e; ++q) {
                int kb = q >> 9;           // 512 halves per K-block
                int r  = q & 511;
                int ln = r >> 4;           // target lane
                int i  = r & 15;           // half index within v16h
                int k  = kb * 32 + ((i >> 3) << 4) + ((ln >> 4) << 3) + (i & 7);
                int n  = ln & 15;
                int c  = k / 9;
                int t9 = k - c * 9;
                int ky = t9 / 3;
                int kx = t9 - ky * 3;
                int hh = row + ky - 1;
                int ww = w0 + n + kx - 1;
                float v = 0.f;
                if ((unsigned)hh < HDIM && (unsigned)ww < WDIM)
                    v = xs[(c * HDIM + hh) * WDIM + ww];
                Blds[q] = (_Float16)v;
            }
            __syncthreads();
            // ---- 16 out-channels x 16 pixels, K = 576 via 18 WMMAs ----
            v8f acc = {};
            #pragma unroll
            for (int kb = 0; kb < KB; ++kb) {
                v16h bfrag = *(const v16h*)(Blds + (kb * 32 + lane) * 16);
                acc = __builtin_amdgcn_wmma_f32_16x16x32_f16(
                    false, afrag[kb], false, bfrag, (short)0, acc, false, false);
            }
            #pragma unroll
            for (int r = 0; r < 8; ++r) {
                int m = mBase + r + (hi << 3);
                convOut[(sL * 128 + m) * 16 + lmod] = acc[r] + bias[r];
            }
        }
        __syncthreads();   // conv chunk complete
        // ---- minGRU recurrence over this chunk (carry in registers) ----
        for (int sL = 0; sL < SCHUNK; ++sL) {
            const int s = chunk * SCHUNK + sL;
            #pragma unroll
            for (int j = 0; j < 4; ++j) {
                int p = tid + j * 256;
                int c = p >> 4, n = p & 15;
                float gate = convOut[(sL * 128 + c) * 16 + n];
                float hidv = convOut[(sL * 128 + 64 + c) * 16 + n];
                float z  = 1.f / (1.f + __expf(-gate));
                float gv = (hidv >= 0.f) ? (hidv + 0.5f) : (1.f / (1.f + __expf(-hidv)));
                hcar[j] = hcar[j] + z * (gv - hcar[j]);
                out[(((size_t)(b * S_TOTAL + s) * HID + c) * HDIM + row) * WDIM + (w0 + n)] = hcar[j];
            }
        }
    }
    // ---- final hidden state ----
    #pragma unroll
    for (int j = 0; j < 4; ++j) {
        int p = tid + j * 256;
        int c = p >> 4, n = p & 15;
        outLast[((b * HID + c) * HDIM + row) * WDIM + (w0 + n)] = hcar[j];
    }
}

extern "C" void kernel_launch(void* const* d_in, const int* in_sizes, int n_in,
                              void* d_out, int out_size, void* d_ws, size_t ws_size,
                              hipStream_t stream) {
    const float* x  = (const float*)d_in[0];
    const float* h0 = (const float*)d_in[1];
    const float* Wc = (const float*)d_in[2];
    const float* bc = (const float*)d_in[3];
    float* out      = (float*)d_out;
    float* outLast  = out + (size_t)4 * S_TOTAL * HID * HDIM * WDIM;

    const size_t shmem = 18432 + 65536;   // B-fragments + conv chunk
    minconv_gru_kernel<<<dim3(1024), dim3(256), shmem, stream>>>(
        x, h0, Wc, bc, out, outLast);
}
